// GNNTowerLite_39831526703908
// MI455X (gfx1250) — compile-verified, matrix operand order checked by reading, not run on hardware
//
#include <hip/hip_runtime.h>
#include <hip/hip_bf16.h>
#include <stdint.h>

typedef __attribute__((ext_vector_type(16))) __bf16 v16bf;
typedef __attribute__((ext_vector_type(8)))  float  v8f;

#define N_NODES 10000
#define IN_DIM  1025
#define KPAD    1088      // 34 * 32  (even k-step count for ping-pong)
#define HID     64
#define NGRAPH  64
#define MTILES  625       // N_NODES / 16

// ---------------- helpers ----------------

__device__ __forceinline__ unsigned short f32_to_bf16_rne(float f) {
    union { float f; unsigned int u; } v; v.f = f;
    unsigned int u = v.u;
    unsigned int r = u + 0x7FFFu + ((u >> 16) & 1u);
    if ((u & 0x7F800000u) == 0x7F800000u) r = u;   // Inf/NaN passthrough
    return (unsigned short)(r >> 16);
}

// 16-bit A/B fragment for v_wmma_*_16x16x32: row-major source, stride in elems.
// Lanes 0-15 : row = lane,    K = {kbase+0..7,  kbase+16..23}
// Lanes 16-31: row = lane-16, K = {kbase+8..15, kbase+24..31}
__device__ __forceinline__ v16bf load_frag(const unsigned short* __restrict__ base,
                                           int row, int stride, int kbase, int lane) {
    const unsigned short* rp = base + (size_t)row * stride + kbase + ((lane >> 4) << 3);
    union { uint4 q[2]; v16bf v; } t;
    t.q[0] = *(const uint4*)(rp);
    t.q[1] = *(const uint4*)(rp + 16);
    return t.v;
}

__device__ __forceinline__ v8f wmma_bf16(v16bf a, v16bf b, v8f c) {
    return __builtin_amdgcn_wmma_f32_16x16x32_bf16(false, a, false, b, (short)0, c,
                                                   false, false);
}

// ---------------- kernels ----------------

__global__ void k_zero(float* __restrict__ p, int n) {
    int i = blockIdx.x * blockDim.x + threadIdx.x;
    if (i < n) p[i] = 0.0f;
}

__global__ void k_cvt_x(const float* __restrict__ x, unsigned short* __restrict__ xb) {
    int i = blockIdx.x * blockDim.x + threadIdx.x;
    if (i >= N_NODES * KPAD) return;
    int n = i / KPAD, k = i - n * KPAD;
    float v = (k < IN_DIM) ? x[(size_t)n * IN_DIM + k] : 0.0f;
    xb[i] = f32_to_bf16_rne(v);
}

__global__ void k_cvt_w0(const float* __restrict__ wself, const float* __restrict__ wneigh,
                         unsigned short* __restrict__ wb) {
    int i = blockIdx.x * blockDim.x + threadIdx.x;
    if (i >= 128 * KPAD) return;
    int r = i / KPAD, k = i - r * KPAD;
    float v = 0.0f;
    if (k < IN_DIM)
        v = (r < HID) ? wself[(size_t)r * IN_DIM + k] : wneigh[(size_t)(r - HID) * IN_DIM + k];
    wb[i] = f32_to_bf16_rne(v);
}

__global__ void k_cvt_w1(const float* __restrict__ wself, const float* __restrict__ wneigh,
                         unsigned short* __restrict__ wb) {
    int i = blockIdx.x * blockDim.x + threadIdx.x;
    if (i >= 128 * HID) return;
    int r = i >> 6, k = i & 63;
    float v = (r < HID) ? wself[r * HID + k] : wneigh[(r - HID) * HID + k];
    wb[i] = f32_to_bf16_rne(v);
}

// P[N,128] = A[N,STRIDE] @ W[128,STRIDE]^T  (bf16 in, f32 accumulate)
// block = 256 threads = 8 waves = 2 M-tiles x 4 column groups (32 cols each).
// Each wave: 16x32 strip, 2 accumulators.
// Ping-pong double buffering: fragment set 1 is loaded BEFORE set 0's WMMAs
// execute (and vice versa), so the two sets' live ranges overlap -> distinct
// registers -> loads of the next k-step overlap the current WMMAs instead of
// an s_wait_loadcnt 0 drain before every matrix op. KSTEPS must be even.
template <int KSTEPS, int STRIDE>
__global__ void k_gemm(const unsigned short* __restrict__ A,
                       const unsigned short* __restrict__ W,
                       float* __restrict__ P) {
    const int lane    = threadIdx.x & 31;
    const int wv      = threadIdx.x >> 5;               // 0..7
    const int mt      = blockIdx.x * 2 + (wv >> 2);
    const int colbase = (wv & 3) * 32;
    if (mt >= MTILES) return;                            // wave-uniform guard
    const int mrow  = mt * 16 + (lane & 15);
    const int wrow0 = colbase + (lane & 15);

    v8f acc0 = {}, acc1 = {};
    v16bf a0, p0, q0, a1, p1, q1;

    a0 = load_frag(A, mrow,       STRIDE, 0, lane);
    p0 = load_frag(W, wrow0,      STRIDE, 0, lane);
    q0 = load_frag(W, wrow0 + 16, STRIDE, 0, lane);

#pragma unroll 1
    for (int kt = 0; kt + 2 < KSTEPS; kt += 2) {
        const int k1 = (kt + 1) * 32;
        const int k2 = (kt + 2) * 32;
        a1 = load_frag(A, mrow,       STRIDE, k1, lane);
        p1 = load_frag(W, wrow0,      STRIDE, k1, lane);
        q1 = load_frag(W, wrow0 + 16, STRIDE, k1, lane);
        acc0 = wmma_bf16(a0, p0, acc0);
        acc1 = wmma_bf16(a0, q0, acc1);
        a0 = load_frag(A, mrow,       STRIDE, k2, lane);
        p0 = load_frag(W, wrow0,      STRIDE, k2, lane);
        q0 = load_frag(W, wrow0 + 16, STRIDE, k2, lane);
        acc0 = wmma_bf16(a1, p1, acc0);
        acc1 = wmma_bf16(a1, q1, acc1);
    }
    {   // tail: buffer 0 holds k-step KSTEPS-2; load final k-step into buffer 1
        const int k1 = (KSTEPS - 1) * 32;
        a1 = load_frag(A, mrow,       STRIDE, k1, lane);
        p1 = load_frag(W, wrow0,      STRIDE, k1, lane);
        q1 = load_frag(W, wrow0 + 16, STRIDE, k1, lane);
        acc0 = wmma_bf16(a0, p0, acc0);
        acc1 = wmma_bf16(a0, q0, acc1);
        acc0 = wmma_bf16(a1, p1, acc0);
        acc1 = wmma_bf16(a1, q1, acc1);
    }

    // C/D layout: VGPR r -> M = r + (lane<16 ? 0 : 8), N = lane&15
    const int rbase = mt * 16 + ((lane >> 4) << 3);
    const int c0    = colbase + (lane & 15);
#pragma unroll
    for (int r = 0; r < 8; ++r) {
        float* row = P + (size_t)(rbase + r) * 128;
        row[c0]      = acc0[r];
        row[c0 + 16] = acc1[r];
    }
}

// one wave per edge: AGG[dst, f] += P[src, 64 + f]  for f in [0,64)
__global__ void k_scatter(const int* __restrict__ ei, int E,
                          const float* __restrict__ P,
                          float* __restrict__ AGG,
                          float* __restrict__ deg, int addDeg) {
    int gt = blockIdx.x * blockDim.x + threadIdx.x;
    int e = gt >> 5, lane = gt & 31;
    if (e >= E) return;
    int s = ei[e];
    int d = ei[E + e];
    if ((unsigned)s >= (unsigned)N_NODES || (unsigned)d >= (unsigned)N_NODES) return;
    const float* y = P + (size_t)s * 128 + 64;
    float* a = AGG + (size_t)d * 64;
    atomicAdd(a + lane,      y[lane]);
    atomicAdd(a + lane + 32, y[lane + 32]);
    if (addDeg && lane == 0) atomicAdd(deg + d, 1.0f);
}

// h0 = relu(self + b_self + agg/deg + b_neigh) -> bf16
__global__ void k_fin0(const float* __restrict__ P0, const float* __restrict__ AGG0,
                       const float* __restrict__ deg,
                       const float* __restrict__ bself, const float* __restrict__ bneigh,
                       unsigned short* __restrict__ hb0) {
    int i = blockIdx.x * blockDim.x + threadIdx.x;
    if (i >= N_NODES * HID) return;
    int n = i >> 6, f = i & 63;
    float dg = deg[n]; dg = dg < 1.0f ? 1.0f : dg;
    float h = P0[(size_t)n * 128 + f] + bself[f] + AGG0[i] / dg + bneigh[f];
    h = h > 0.0f ? h : 0.0f;
    hb0[i] = f32_to_bf16_rne(h);
}

// h1 = relu(...); scatter-add into per-graph accumulators
__global__ void k_fin1(const float* __restrict__ P1, const float* __restrict__ AGG1,
                       const float* __restrict__ deg,
                       const float* __restrict__ bself, const float* __restrict__ bneigh,
                       const int* __restrict__ bvec,
                       float* __restrict__ outsum, float* __restrict__ cnt) {
    int i = blockIdx.x * blockDim.x + threadIdx.x;
    if (i >= N_NODES * HID) return;
    int n = i >> 6, f = i & 63;
    float dg = deg[n]; dg = dg < 1.0f ? 1.0f : dg;
    float h = P1[(size_t)n * 128 + f] + bself[f] + AGG1[i] / dg + bneigh[f];
    h = h > 0.0f ? h : 0.0f;
    int b = bvec[n];
    if ((unsigned)b < (unsigned)NGRAPH) {
        atomicAdd(&outsum[b * HID + f], h);
        if (f == 0) atomicAdd(&cnt[b], 1.0f);
    }
}

__global__ void k_div(const float* __restrict__ outsum, const float* __restrict__ cnt,
                      float* __restrict__ out) {
    int i = blockIdx.x * blockDim.x + threadIdx.x;
    if (i >= NGRAPH * HID) return;
    float c = cnt[i >> 6]; c = c < 1.0f ? 1.0f : c;
    out[i] = outsum[i] / c;
}

// ---------------- launch ----------------

static inline size_t alignup(size_t v) { return (v + 255) & ~(size_t)255; }

extern "C" void kernel_launch(void* const* d_in, const int* in_sizes, int n_in,
                              void* d_out, int out_size, void* d_ws, size_t ws_size,
                              hipStream_t stream) {
    const float* x        = (const float*)d_in[0];
    const int*   ei       = (const int*)d_in[1];
    const int*   bvec     = (const int*)d_in[2];
    // d_in[3] = batch_size (device scalar); NGRAPH hardcoded per reference
    const float* w_self0  = (const float*)d_in[4];
    const float* b_self0  = (const float*)d_in[5];
    const float* w_neigh0 = (const float*)d_in[6];
    const float* b_neigh0 = (const float*)d_in[7];
    const float* w_self1  = (const float*)d_in[8];
    const float* b_self1  = (const float*)d_in[9];
    const float* w_neigh1 = (const float*)d_in[10];
    const float* b_neigh1 = (const float*)d_in[11];
    const int E = in_sizes[1] / 2;

    // workspace layout: [zero region | xb | wb0 | P0 | hb0 | wb1 | P1]
    char* p = (char*)d_ws;
    float* deg    = (float*)p;
    float* AGG0   = deg + N_NODES;
    float* AGG1   = AGG0 + (size_t)N_NODES * HID;
    float* outsum = AGG1 + (size_t)N_NODES * HID;
    float* cnt    = outsum + NGRAPH * HID;
    const int ZN  = N_NODES + 2 * N_NODES * HID + NGRAPH * HID + NGRAPH;
    p += alignup((size_t)ZN * sizeof(float));
    unsigned short* xb  = (unsigned short*)p; p += alignup((size_t)N_NODES * KPAD * 2);
    unsigned short* wb0 = (unsigned short*)p; p += alignup((size_t)128 * KPAD * 2);
    float*          P0  = (float*)p;          p += alignup((size_t)N_NODES * 128 * 4);
    unsigned short* hb0 = (unsigned short*)p; p += alignup((size_t)N_NODES * HID * 2);
    unsigned short* wb1 = (unsigned short*)p; p += alignup((size_t)128 * HID * 2);
    float*          P1  = (float*)p;

    float* out = (float*)d_out;

    const int T = 256;
    const int GEMM_BLOCKS = (MTILES + 1) / 2;   // 2 M-tiles per block
    // 1. zero accumulators (must be re-zeroed every call: graph replay, no re-poison)
    k_zero<<<(ZN + T - 1) / T, T, 0, stream>>>(deg, ZN);
    // 2. bf16 conversions (padded K)
    k_cvt_x <<<(N_NODES * KPAD + T - 1) / T, T, 0, stream>>>(x, xb);
    k_cvt_w0<<<(128 * KPAD + T - 1) / T, T, 0, stream>>>(w_self0, w_neigh0, wb0);
    k_cvt_w1<<<(128 * HID + T - 1) / T, T, 0, stream>>>(w_self1, w_neigh1, wb1);
    // 3. layer-0 fused GEMM: P0[:, :64] = x@Wself.T ; P0[:, 64:] = x@Wneigh.T
    k_gemm<KPAD / 32, KPAD><<<GEMM_BLOCKS, T, 0, stream>>>(xb, wb0, P0);
    // 4. edge scatter of projected neighbor features + degree
    k_scatter<<<((size_t)E * 32 + T - 1) / T, T, 0, stream>>>(ei, E, P0, AGG0, deg, 1);
    // 5. finalize layer 0 -> bf16 h0
    k_fin0<<<(N_NODES * HID + T - 1) / T, T, 0, stream>>>(P0, AGG0, deg, b_self0, b_neigh0, hb0);
    // 6. layer-1 fused GEMM
    k_gemm<HID / 32, HID><<<GEMM_BLOCKS, T, 0, stream>>>(hb0, wb1, P1);
    // 7. edge scatter (reuse degree)
    k_scatter<<<((size_t)E * 32 + T - 1) / T, T, 0, stream>>>(ei, E, P1, AGG1, deg, 0);
    // 8. finalize layer 1 + per-graph readout accumulation
    k_fin1<<<(N_NODES * HID + T - 1) / T, T, 0, stream>>>(P1, AGG1, deg, b_self1, b_neigh1,
                                                          bvec, outsum, cnt);
    // 9. mean divide into output [64, 64]
    k_div<<<(NGRAPH * HID + T - 1) / T, T, 0, stream>>>(outsum, cnt, out);
}
